// MultiheadAttentionZero_53437983097003
// MI455X (gfx1250) — matrix operand order repaired and checked by reading, not compiled, not run on hardware
//
#include <hip/hip_runtime.h>
#include <hip/hip_bf16.h>

// ---------------------------------------------------------------------------
// MHA forward for MI455X (gfx1250, wave32, WMMA + async global->LDS DMA).
//   B=4, T=S=2048, DIM=1024, HEADS=16, HEAD_DIM=64
// Pipeline: f32->bf16 convert -> WMMA GEMM (async-LDS B tiles, 32x64 wave
// tiles, fused RoPE) -> flash attention (async K tiles, WMMA QK^T / PV)
//           -> WMMA output projection (f32 out).
// ---------------------------------------------------------------------------

typedef __bf16 bf16_t;
typedef __attribute__((ext_vector_type(16))) __bf16 v16bf;
typedef __attribute__((ext_vector_type(8)))  __bf16 v8bf;
typedef __attribute__((ext_vector_type(8)))  float  v8f;

#define DIMN   1024
#define HEADS  16
#define HD     64
#define BB     4
#define TT     2048
#define SSN    2048
#define MROWS  (BB * TT)      // 8192 rows for every GEMM

// ---- CDNA5 async global->LDS copy (ASYNCcnt-tracked DMA) ------------------
__device__ __forceinline__ unsigned ldsoff(const void* p) {
  return (unsigned)(uintptr_t)p;          // low 32 bits of flat addr = LDS off
}
__device__ __forceinline__ void async_ld_b128(unsigned lds, const void* gptr) {
  asm volatile("global_load_async_to_lds_b128 %0, %1, off"
               :: "v"(lds), "v"(gptr) : "memory");
}
__device__ __forceinline__ void wait_async_le1() {
  asm volatile("s_wait_asynccnt 0x1" ::: "memory");
}
__device__ __forceinline__ void wait_async0() {
  asm volatile("s_wait_asynccnt 0x0" ::: "memory");
}

// ---- fragment loaders (CDNA5 16-bit WMMA VGPR layouts) --------------------
// A 16x32: lane m = l%16; chunks K = k0..k0+7 and k0+16..k0+23, k0=(l/16)*8
// B 32x16: lane n = l%16; chunks K = k0..k0+7 and k0+16..k0+23, k0=(l/16)*8
__device__ __forceinline__ v16bf ldpair(const bf16_t* p, int gap) {
  v8bf lo = *(const v8bf*)p;
  v8bf hi = *(const v8bf*)(p + gap);
  v16bf r;
#pragma unroll
  for (int i = 0; i < 8; ++i) { r[i] = lo[i]; r[i + 8] = hi[i]; }
  return r;
}
__device__ __forceinline__ v16bf ldA(const bf16_t* p) { return ldpair(p, 16); }
__device__ __forceinline__ v16bf ldB(const bf16_t* p) { return ldpair(p, 16); }

__device__ __forceinline__ v8f wmma_bf16(v16bf a, v16bf b, v8f c) {
  return __builtin_amdgcn_wmma_f32_16x16x32_bf16(false, a, false, b,
                                                 (short)0, c, false, false);
}

// ---------------------------------------------------------------------------
// f32 -> bf16 conversion (n multiple of 2048)
// ---------------------------------------------------------------------------
__global__ __launch_bounds__(256) void cvt_bf16_kernel(
    const float* __restrict__ src, bf16_t* __restrict__ dst, int n) {
  int i = (blockIdx.x * 256 + threadIdx.x) * 8;
  if (i + 8 <= n) {
    float4 a = *(const float4*)(src + i);
    float4 b = *(const float4*)(src + i + 4);
    v8bf o;
    o[0] = (bf16_t)a.x; o[1] = (bf16_t)a.y; o[2] = (bf16_t)a.z; o[3] = (bf16_t)a.w;
    o[4] = (bf16_t)b.x; o[5] = (bf16_t)b.y; o[6] = (bf16_t)b.z; o[7] = (bf16_t)b.w;
    *(v8bf*)(dst + i) = o;
  }
}

// ---------------------------------------------------------------------------
// GEMM: out = X(M,1024) @ W(1024,1024)^T  (W row-major: [out_ch][k])
// block = 8 waves; each wave one 32(M) x 64(N) tile (two M sub-tiles reuse
// every B fragment); all waves share the 64(N) x 32(K) B tile, staged in LDS
// by async DMA, double-buffered.
// mode 0: f32 row-major out (final projection)
// mode 1: RoPE epilogue (shfl_xor pairing), bf16 out in (B,H,T,64)
// mode 2: no RoPE, bf16 out in (B,H,S,64)  (V projection)
// ---------------------------------------------------------------------------
__global__ __launch_bounds__(256) void gemm_bf16_kernel(
    const bf16_t* __restrict__ X, const bf16_t* __restrict__ W,
    const float* __restrict__ freq, bf16_t* __restrict__ outB,
    float* __restrict__ outF, int mode) {
  __shared__ __align__(16) bf16_t Bt[2][64][40];   // [buf][n][k] k-pad 40

  const int lane = threadIdx.x & 31;
  const int wave = threadIdx.x >> 5;
  const int half = lane >> 4;
  const int lcol = lane & 15;
  const int n0 = blockIdx.x * 64;
  const int m0 = blockIdx.y * 256 + wave * 32;

  const bf16_t* Arow0 = X + (size_t)(m0 + lcol) * DIMN;
  const bf16_t* Arow1 = X + (size_t)(m0 + 16 + lcol) * DIMN;

  // per-thread async staging slot: one b128 (8 bf16) of the 64x32 B tile
  const int brow = threadIdx.x >> 2;        // 0..63  (= n)
  const int bcol = (threadIdx.x & 3) * 8;   // 0,8,16,24 (= k)
  const bf16_t* wsrc = W + (size_t)(n0 + brow) * DIMN + bcol;

  async_ld_b128(ldsoff(&Bt[0][brow][bcol]), wsrc);        // stage kk = 0
  v16bf a0_cur = ldA(Arow0 + half * 8);
  v16bf a1_cur = ldA(Arow1 + half * 8);

  v8f acc[2][4] = {};
  int buf = 0;
  for (int kk = 0; kk < DIMN; kk += 32) {
    v16bf a0_nxt = a0_cur, a1_nxt = a1_cur;
    const bool more = (kk + 32) < DIMN;
    if (more) {
      async_ld_b128(ldsoff(&Bt[buf ^ 1][brow][bcol]), wsrc + kk + 32);
      a0_nxt = ldA(Arow0 + kk + 32 + half * 8);
      a1_nxt = ldA(Arow1 + kk + 32 + half * 8);
    }
    if (more) wait_async_le1(); else wait_async0();   // current tile ready
    __syncthreads();
#pragma unroll
    for (int jn = 0; jn < 4; ++jn) {
      v16bf bfrag = ldB(&Bt[buf][jn * 16 + lcol][half * 8]);
      acc[0][jn] = wmma_bf16(a0_cur, bfrag, acc[0][jn]);
      acc[1][jn] = wmma_bf16(a1_cur, bfrag, acc[1][jn]);
    }
    a0_cur = a0_nxt;
    a1_cur = a1_nxt;
    buf ^= 1;
    __syncthreads();          // reads done before next overwrite of buf^1
  }

  if (mode == 0) {
#pragma unroll
    for (int mi = 0; mi < 2; ++mi)
#pragma unroll
      for (int jn = 0; jn < 4; ++jn)
#pragma unroll
        for (int r = 0; r < 8; ++r) {
          int gm = m0 + mi * 16 + r + 8 * half;
          int c  = n0 + jn * 16 + lcol;
          outF[(size_t)gm * DIMN + c] = acc[mi][jn][r];
        }
    return;
  }

#pragma unroll
  for (int mi = 0; mi < 2; ++mi)
#pragma unroll
    for (int jn = 0; jn < 4; ++jn) {
      int c = n0 + jn * 16 + lcol;
      int h = c >> 6, dd = c & 63;
      int fidx = dd >> 1;
#pragma unroll
      for (int r = 0; r < 8; ++r) {
        int gm = m0 + mi * 16 + r + 8 * half;
        int b = gm >> 11, t = gm & (TT - 1);
        float v = acc[mi][jn][r];
        float res = v;
        if (mode == 1) {
          float o = __shfl_xor(v, 1);     // rotation partner column (c^1)
          float f = freq[t * (HD / 2) + fidx];
          float sn, cs;
          __sincosf(f, &sn, &cs);
          res = (dd & 1) ? (o * sn + v * cs) : (v * cs - o * sn);
        }
        outB[(((size_t)(b * HEADS + h) * TT) + t) * HD + dd] = (bf16_t)res;
      }
    }
}

// ---------------------------------------------------------------------------
// Flash attention per (b,h). Block: 8 waves x 16 queries = 128 rows.
// K tile staged by async global->LDS DMA; V tile loaded + transposed by VALU
// (overlaps the DMA); P transposed through wave-private LDS tile.
// ---------------------------------------------------------------------------
__global__ __launch_bounds__(256) void attn_kernel(
    const bf16_t* __restrict__ Qr, const bf16_t* __restrict__ Kr,
    const bf16_t* __restrict__ Vr, const int* __restrict__ kmask,
    bf16_t* __restrict__ AO) {
  __shared__ __align__(16) bf16_t Kt[64][72];      // K tile   [s][d]
  __shared__ __align__(16) bf16_t Vt[64][72];      // V^T tile [d][s]
  __shared__ __align__(16) bf16_t Pt[8][16][72];   // per-wave P tile [m][s]

  const int lane = threadIdx.x & 31;
  const int wave = threadIdx.x >> 5;
  const int half = lane >> 4;
  const int lcol = lane & 15;
  const int bh = blockIdx.y;
  const int b = bh >> 4, h = bh & 15;
  const int t0 = blockIdx.x * 128 + wave * 16;
  const size_t base = (size_t)bh * SSN * HD;
  const float K2 = 0.125f * 1.44269504f;           // scale * log2(e)

  const bf16_t* qrow = Qr + base + (size_t)(t0 + lcol) * HD;
  v16bf aq0 = ldA(qrow + half * 8);
  v16bf aq1 = ldA(qrow + 32 + half * 8);

  float run_m[8], run_s[8];
  v8f acc[4] = {};
#pragma unroll
  for (int r = 0; r < 8; ++r) { run_m[r] = -3.0e38f; run_s[r] = 0.0f; }

  for (int s0 = 0; s0 < SSN; s0 += 64) {
    // K tile: async DMA straight into padded LDS rows (2 x b128 per thread)
#pragma unroll
    for (int it = 0; it < 2; ++it) {
      int idx = threadIdx.x + it * 256;
      int row = idx >> 3;
      int col = (idx & 7) << 3;
      async_ld_b128(ldsoff(&Kt[row][col]),
                    Kr + base + (size_t)(s0 + row) * HD + col);
    }
    // V tile: regular load + transposed LDS store (overlaps async K DMA)
#pragma unroll
    for (int it = 0; it < 2; ++it) {
      int idx = threadIdx.x + it * 256;
      int row = idx >> 3;
      int col = (idx & 7) << 3;
      const bf16_t* vp = Vr + base + (size_t)(s0 + row) * HD + col;
      v8bf vv = *(const v8bf*)vp;
      if (s0 + 64 < SSN) __builtin_prefetch(vp + 64 * HD, 0, 3);
#pragma unroll
      for (int i = 0; i < 8; ++i) Vt[col + i][row] = vv[i];
    }
    wait_async0();
    __syncthreads();

    // scores = Q @ K^T   (16 x 64, 8 WMMAs)
    v8f sc[4] = {};
#pragma unroll
    for (int jn = 0; jn < 4; ++jn) {
      const bf16_t* kr = &Kt[jn * 16 + lcol][half * 8];
      sc[jn] = wmma_bf16(aq0, ldB(kr), sc[jn]);
      sc[jn] = wmma_bf16(aq1, ldB(kr + 32), sc[jn]);
    }

    int ok[4];
#pragma unroll
    for (int jn = 0; jn < 4; ++jn)
      ok[jn] = kmask[b * SSN + s0 + jn * 16 + lcol];

    // online softmax (rows live in 16-lane half-waves -> butterfly reduce)
#pragma unroll
    for (int r = 0; r < 8; ++r) {
      float x[4];
#pragma unroll
      for (int jn = 0; jn < 4; ++jn)
        x[jn] = ok[jn] ? sc[jn][r] * K2 : -1.0e9f;
      float cm = fmaxf(fmaxf(x[0], x[1]), fmaxf(x[2], x[3]));
#pragma unroll
      for (int off = 1; off < 16; off <<= 1)
        cm = fmaxf(cm, __shfl_xor(cm, off));
      float mnew = fmaxf(run_m[r], cm);
      float corr = exp2f(run_m[r] - mnew);
      float psum = 0.0f;
#pragma unroll
      for (int jn = 0; jn < 4; ++jn) {
        float p = exp2f(x[jn] - mnew);
        Pt[wave][r + 8 * half][jn * 16 + lcol] = (bf16_t)p;
        psum += p;
      }
#pragma unroll
      for (int off = 1; off < 16; off <<= 1) psum += __shfl_xor(psum, off);
      run_s[r] = run_s[r] * corr + psum;
      run_m[r] = mnew;
#pragma unroll
      for (int jn = 0; jn < 4; ++jn) acc[jn][r] *= corr;
    }

    // O += P @ V   (A from wave-private LDS P, B from transposed V tile)
    const bf16_t* prow = &Pt[wave][lcol][0];
#pragma unroll
    for (int ks = 0; ks < 2; ++ks) {
      v16bf ap = ldA(prow + ks * 32 + half * 8);
#pragma unroll
      for (int jn = 0; jn < 4; ++jn) {
        const bf16_t* vp = &Vt[jn * 16 + lcol][ks * 32 + half * 8];
        acc[jn] = wmma_bf16(ap, ldB(vp), acc[jn]);
      }
    }
    __syncthreads();
  }

  // normalize and write bf16 out in (B, T, DIM) layout at column h*64
#pragma unroll
  for (int r = 0; r < 8; ++r) {
    float inv = 1.0f / run_s[r];
    int t = t0 + r + 8 * half;
#pragma unroll
    for (int jn = 0; jn < 4; ++jn) {
      int c = h * HD + jn * 16 + lcol;
      AO[((size_t)(b * TT + t)) * DIMN + c] = (bf16_t)(acc[jn][r] * inv);
    }
  }
}

// ---------------------------------------------------------------------------
extern "C" void kernel_launch(void* const* d_in, const int* in_sizes, int n_in,
                              void* d_out, int out_size, void* d_ws,
                              size_t ws_size, hipStream_t stream) {
  (void)in_sizes; (void)n_in; (void)out_size; (void)ws_size;
  const float* q_x     = (const float*)d_in[0];
  const float* kv_x    = (const float*)d_in[1];
  const float* freq_q  = (const float*)d_in[2];
  const float* freq_kv = (const float*)d_in[3];
  const int*   kmask   = (const int*)d_in[4];
  const float* Wq = (const float*)d_in[5];
  const float* Wk = (const float*)d_in[6];
  const float* Wv = (const float*)d_in[7];
  const float* Wp = (const float*)d_in[8];
  float* out = (float*)d_out;

  char* ws = (char*)d_ws;
  size_t off = 0;
  auto carve = [&](size_t bytes) {
    void* p = ws + off;
    off += (bytes + 255) & ~(size_t)255;
    return p;
  };
  const size_t WSZ = (size_t)DIMN * DIMN * sizeof(bf16_t);   // 2 MB
  const size_t ASZ = (size_t)MROWS * DIMN * sizeof(bf16_t);  // 16 MB
  bf16_t* WqB  = (bf16_t*)carve(WSZ);
  bf16_t* WkB  = (bf16_t*)carve(WSZ);
  bf16_t* WvB  = (bf16_t*)carve(WSZ);
  bf16_t* WpB  = (bf16_t*)carve(WSZ);
  bf16_t* XqB  = (bf16_t*)carve(ASZ);
  bf16_t* XkvB = (bf16_t*)carve(ASZ);
  bf16_t* QrB  = (bf16_t*)carve(ASZ);   // (B,H,T,64)
  bf16_t* KrB  = (bf16_t*)carve(ASZ);   // (B,H,S,64)
  bf16_t* VrB  = (bf16_t*)carve(ASZ);   // (B,H,S,64)
  bf16_t* AOB  = (bf16_t*)carve(ASZ);   // (B,T,DIM)

  const int wN = DIMN * DIMN;    // 1,048,576
  const int aN = MROWS * DIMN;   // 8,388,608
  cvt_bf16_kernel<<<wN / 2048, 256, 0, stream>>>(Wq, WqB, wN);
  cvt_bf16_kernel<<<wN / 2048, 256, 0, stream>>>(Wk, WkB, wN);
  cvt_bf16_kernel<<<wN / 2048, 256, 0, stream>>>(Wv, WvB, wN);
  cvt_bf16_kernel<<<wN / 2048, 256, 0, stream>>>(Wp, WpB, wN);
  cvt_bf16_kernel<<<aN / 2048, 256, 0, stream>>>(q_x, XqB, aN);
  cvt_bf16_kernel<<<aN / 2048, 256, 0, stream>>>(kv_x, XkvB, aN);

  dim3 gg(DIMN / 64, MROWS / 256);   // (16, 32)
  gemm_bf16_kernel<<<gg, 256, 0, stream>>>(XqB,  WqB, freq_q,  QrB, nullptr, 1);
  gemm_bf16_kernel<<<gg, 256, 0, stream>>>(XkvB, WkB, freq_kv, KrB, nullptr, 1);
  gemm_bf16_kernel<<<gg, 256, 0, stream>>>(XkvB, WvB, nullptr, VrB, nullptr, 2);

  dim3 ga(TT / 128, BB * HEADS);     // (16, 64)
  attn_kernel<<<ga, 256, 0, stream>>>(QrB, KrB, VrB, kmask, AOB);

  gemm_bf16_kernel<<<gg, 256, 0, stream>>>(AOB, WpB, nullptr, nullptr, out, 0);
}
